// CustomCLIPAttention_40407052321204
// MI455X (gfx1250) — matrix operand order repaired and checked by reading, not compiled, not run on hardware
//
#include <hip/hip_runtime.h>
#include <hip/hip_bf16.h>

// ---------------------------------------------------------------------------
// CLIP attention for MI455X (gfx1250, wave32, WMMA).
// B=32, S=577, E=1024, H=16, D=64.
// f32 -> f16 conversion, all GEMMs + attention on v_wmma_f32_16x16x32_f16
// (f32 accumulate). Layouts chosen so every WMMA operand fragment is a
// contiguous 16-32B per-lane global/LDS load (no transposes needed except
// the P-matrix LDS bounce in attention, which doubles as the softmax pass).
// ---------------------------------------------------------------------------

#define BB   32
#define SS   577
#define EE   1024
#define HH   16
#define DD   64
#define NN   (BB * SS)        // 18464 rows = 1154 * 16 exactly
#define MT   (NN / 16)        // 1154 M tiles
#define SPQ  592              // S padded to 16  (37 q-tiles)
#define SPK  608              // S padded to 32  (38 k-tiles of 16, 19 of 32)
#define NTW  8                // 8 N-tiles of 16 per wave -> 16x128 wave tile

typedef __attribute__((ext_vector_type(16))) _Float16 v16h;
typedef __attribute__((ext_vector_type(8)))  _Float16 v8h;
typedef __attribute__((ext_vector_type(8)))  float    v8f;

union V16 { v16h v; v8h h[2]; };

static __device__ __forceinline__ v8h ld8(const _Float16* p) {
    return *(const v8h*)p;
}

static __device__ __forceinline__ v8f wmma16(const V16& a, const V16& b, v8f c) {
    // D = A(16x32 f16) * B(32x16 f16) + C(16x16 f32)
    return __builtin_amdgcn_wmma_f32_16x16x32_f16(
        /*neg_a=*/false, a.v, /*neg_b=*/false, b.v,
        /*c_mod=*/(short)0, c, /*reuse_a=*/false, /*reuse_b=*/false);
}

// ---------------------------------------------------------------------------
// f32 -> f16 elementwise conversion (grid-stride, 4 elems/thread)
// ---------------------------------------------------------------------------
__global__ void cvt_f32_f16(const float* __restrict__ src,
                            _Float16* __restrict__ dst, long n4) {
    long i = (long)blockIdx.x * blockDim.x + threadIdx.x;
    long stride = (long)gridDim.x * blockDim.x;
    for (; i < n4; i += stride) {
        const float4 v = ((const float4*)src)[i];
        v8h* d = (v8h*)dst;  // unused; keep types simple below
        (void)d;
        _Float16 h0 = (_Float16)v.x, h1 = (_Float16)v.y;
        _Float16 h2 = (_Float16)v.z, h3 = (_Float16)v.w;
        dst[i * 4 + 0] = h0; dst[i * 4 + 1] = h1;
        dst[i * 4 + 2] = h2; dst[i * 4 + 3] = h3;
    }
}

// ---------------------------------------------------------------------------
// Projection GEMM: Y[n,f] = sum_e A16[n,e] * W16[f,e] + bias[f]
// One wave computes a 16(M) x 128(N) tile (8 WMMA accumulators).
// mode selects the store path:
//   0: Q  -> q16 [bh][SPQ][64]  (scaled by D^-0.5)
//   1: K  -> k16 [bh][SPK][64]
//   2: V  -> vt16[bh][64][SPK]  (transposed for the PV WMMA B-fragment)
//   3: O  -> out f32 [n][E]
// ---------------------------------------------------------------------------
__global__ void gemm_proj(const _Float16* __restrict__ A,
                          const _Float16* __restrict__ W,
                          const float* __restrict__ bias,
                          void* __restrict__ dst, int mode) {
    const int lane = threadIdx.x;       // 0..31
    const int half = lane >> 4;         // which 16-lane half
    const int l15  = lane & 15;
    const int mt   = blockIdx.x * blockDim.y + threadIdx.y;  // M tile
    const int nt   = blockIdx.y;                             // 128-wide N tile
    if (mt >= MT) return;               // wave-uniform guard (EXEC stays full)

    const _Float16* arow = A + (long)(mt * 16 + l15) * EE;   // A row = lane&15
    const _Float16* wbase = W + (long)(nt * 128 + l15) * EE + half * 16;

    v8f acc[NTW] = {};
    for (int k0 = 0; k0 < EE; k0 += 32) {
        V16 a;                                   // A 16x32: 16 contiguous K per half
        a.h[0] = ld8(arow + k0 + half * 16);     // K {0..7} / {16..23}
        a.h[1] = ld8(arow + k0 + half * 16 + 8); // K {8..15} / {24..31}
#pragma unroll
        for (int t = 0; t < NTW; ++t) {
            const _Float16* wrow = wbase + (long)(t * 16) * EE + k0;
            V16 b;                               // B 32x16: 16 contiguous K
            b.h[0] = ld8(wrow);
            b.h[1] = ld8(wrow + 8);
            acc[t] = wmma16(a, b, acc[t]);
        }
    }

    // hoist biases (one per N sub-tile column owned by this lane)
    float bv[NTW];
#pragma unroll
    for (int t = 0; t < NTW; ++t) bv[t] = bias[nt * 128 + t * 16 + l15];

#pragma unroll
    for (int i = 0; i < 8; ++i) {                // C row = i + 8*half
        const int n = mt * 16 + i + half * 8;
        const int b_ = n / SS, s = n % SS;       // computed once per row
#pragma unroll
        for (int t = 0; t < NTW; ++t) {
            const int f  = nt * 128 + t * 16 + l15;   // C col = lane&15
            const float y = acc[t][i] + bv[t];
            const int hh = f >> 6, d = f & 63;
            const long bh = (long)b_ * HH + hh;
            if (mode == 0) {
                ((_Float16*)dst)[(bh * SPQ + s) * 64 + d] = (_Float16)(y * 0.125f);
            } else if (mode == 1) {
                ((_Float16*)dst)[(bh * SPK + s) * 64 + d] = (_Float16)y;
            } else if (mode == 2) {
                ((_Float16*)dst)[(bh * 64 + d) * SPK + s] = (_Float16)y;
            } else {
                ((float*)dst)[(long)n * EE + f] = y;
            }
        }
    }
}

// ---------------------------------------------------------------------------
// Attention: one wave per (batch*head, 16-query tile).
//   pass 1: scores = Q*K^T (WMMA), row-max via shfl butterflies, raw f16
//           scores parked in LDS (this is the C-layout -> A-layout transpose)
//   pass 2: A-fragments of exp(score - rowmax) from LDS, PV WMMA against
//           V^T, row-sum, normalize at the context store.
// ---------------------------------------------------------------------------
__global__ void attn_kernel(const _Float16* __restrict__ Q,
                            const _Float16* __restrict__ K,
                            const _Float16* __restrict__ VT,
                            _Float16* __restrict__ ctx) {
    __shared__ _Float16 lds_p[16][SPK];     // raw scores, then probs source
    __shared__ float lds_rm[16];
    __shared__ float lds_sum[16];

    const int lane = threadIdx.x;
    const int half = lane >> 4;
    const int l15  = lane & 15;
    const int qt   = blockIdx.x;            // 0..36
    const int bh   = blockIdx.y;            // 0..511
    const long headQ = (long)bh * SPQ * 64;
    const long headK = (long)bh * SPK * 64;
    const long headV = (long)bh * 64 * SPK;

    // Q A-fragments for the two 32-wide halves of D=64 (held all kernel)
    const _Float16* qrow = Q + headQ + (long)(qt * 16 + l15) * 64;
    V16 aq0, aq1;
    aq0.h[0] = ld8(qrow + 0  + half * 16);
    aq0.h[1] = ld8(qrow + 8  + half * 16);
    aq1.h[0] = ld8(qrow + 32 + half * 16);
    aq1.h[1] = ld8(qrow + 40 + half * 16);

    const float NEG_INF = -__builtin_inff();
    float lm[8];
#pragma unroll
    for (int i = 0; i < 8; ++i) lm[i] = NEG_INF;

    // ---- pass 1: scores -> LDS, track per-lane row maxes ----
    for (int kt = 0; kt < SPK / 16; ++kt) {
        const int kxcol = kt * 16 + l15;                 // score column
        const _Float16* kb = K + headK + (long)kxcol * 64;
        V16 b0, b1;
        b0.h[0] = ld8(kb + half * 16);
        b0.h[1] = ld8(kb + half * 16 + 8);
        b1.h[0] = ld8(kb + 32 + half * 16);
        b1.h[1] = ld8(kb + 32 + half * 16 + 8);
        v8f c = {};
        c = wmma16(aq0, b0, c);
        c = wmma16(aq1, b1, c);
        const bool valid = kxcol < SS;
#pragma unroll
        for (int i = 0; i < 8; ++i) {                    // row = i + 8*half
            const float sc = valid ? c[i] : NEG_INF;     // mask K padding
            lm[i] = fmaxf(lm[i], sc);
            lds_p[i + half * 8][kxcol] = (_Float16)sc;
        }
    }
    // row max: butterfly across the 16 lanes of each half
#pragma unroll
    for (int off = 1; off < 16; off <<= 1) {
#pragma unroll
        for (int i = 0; i < 8; ++i)
            lm[i] = fmaxf(lm[i], __shfl_xor(lm[i], off, 32));
    }
    if (l15 == 0) {
#pragma unroll
        for (int i = 0; i < 8; ++i) lds_rm[half * 8 + i] = lm[i];
    }
    __syncthreads();

    // ---- pass 2: exp + PV ----
    const float rm = lds_rm[l15];           // A-layout row = lane&15
    float psum = 0.f;
    v8f out[4] = {};
    for (int kt2 = 0; kt2 < SPK / 32; ++kt2) {
        const int k0 = kt2 * 32;
        V16 ap;                             // probs A 16x32 straight from LDS
        ap.h[0] = *(const v8h*)&lds_p[l15][k0 + half * 16];
        ap.h[1] = *(const v8h*)&lds_p[l15][k0 + half * 16 + 8];
#pragma unroll
        for (int j = 0; j < 16; ++j) {
            const float s = (float)ap.v[j];
            const float p = __expf(s - rm); // pad cols were -inf -> p = 0
            psum += p;
            ap.v[j] = (_Float16)p;
        }
#pragma unroll
        for (int t = 0; t < 4; ++t) {       // B 32x16 from V^T: contiguous K
            const _Float16* vrow =
                VT + headV + (long)(t * 16 + l15) * SPK + k0 + half * 16;
            V16 b;
            b.h[0] = ld8(vrow);
            b.h[1] = ld8(vrow + 8);
            out[t] = wmma16(ap, b, out[t]);
        }
    }
    psum += __shfl_xor(psum, 16, 32);       // join the two K halves of a row
    if (half == 0) lds_sum[l15] = psum;
    __syncthreads();

    // ---- normalize + store context [B,S,E] as f16 ----
    const int b_ = bh / HH, hh = bh % HH;
#pragma unroll
    for (int i = 0; i < 8; ++i) {
        const int r = i + half * 8;
        const int s = qt * 16 + r;
        if (s < SS) {
            const float inv = 1.0f / lds_sum[r];
#pragma unroll
            for (int t = 0; t < 4; ++t) {
                ctx[((long)(b_ * SS + s)) * EE + hh * 64 + t * 16 + l15] =
                    (_Float16)(out[t][i] * inv);
            }
        }
    }
}

// ---------------------------------------------------------------------------
// Host side
// ---------------------------------------------------------------------------
static inline size_t align256(size_t x) { return (x + 255) & ~(size_t)255; }

extern "C" void kernel_launch(void* const* d_in, const int* in_sizes, int n_in,
                              void* d_out, int out_size, void* d_ws, size_t ws_size,
                              hipStream_t stream) {
    (void)in_sizes; (void)n_in; (void)out_size; (void)ws_size;

    const float* x  = (const float*)d_in[0];
    const float* qw = (const float*)d_in[1];
    const float* qb = (const float*)d_in[2];
    const float* kw = (const float*)d_in[3];
    const float* kb = (const float*)d_in[4];
    const float* vw = (const float*)d_in[5];
    const float* vb = (const float*)d_in[6];
    const float* ow = (const float*)d_in[7];
    const float* ob = (const float*)d_in[8];

    // workspace carve-up
    char* ws = (char*)d_ws;
    size_t off = 0;
    const size_t X16_B  = (size_t)NN * EE * 2;
    const size_t W16_B  = (size_t)EE * EE * 2;
    const size_t Q16_B  = (size_t)BB * HH * SPQ * DD * 2;
    const size_t K16_B  = (size_t)BB * HH * SPK * DD * 2;
    const size_t VT16_B = (size_t)BB * HH * DD * SPK * 2;
    const size_t CTX_B  = (size_t)NN * EE * 2;

    _Float16* X16  = (_Float16*)(ws + off); off = align256(off + X16_B);
    _Float16* W16q = (_Float16*)(ws + off); off = align256(off + W16_B);
    _Float16* W16k = (_Float16*)(ws + off); off = align256(off + W16_B);
    _Float16* W16v = (_Float16*)(ws + off); off = align256(off + W16_B);
    _Float16* W16o = (_Float16*)(ws + off); off = align256(off + W16_B);
    _Float16* Q16  = (_Float16*)(ws + off); off = align256(off + Q16_B);
    _Float16* K16  = (_Float16*)(ws + off); off = align256(off + K16_B);
    _Float16* VT16 = (_Float16*)(ws + off); off = align256(off + VT16_B);
    _Float16* CTX  = (_Float16*)(ws + off); off = align256(off + CTX_B);

    // zero the padded Q/K/V buffers so pad rows/cols feed zeros to the WMMAs
    hipMemsetAsync(Q16, 0, Q16_B, stream);
    hipMemsetAsync(K16, 0, K16_B, stream);
    hipMemsetAsync(VT16, 0, VT16_B, stream);

    // f32 -> f16 conversions (vectorized x4; all sizes divisible by 4)
    cvt_f32_f16<<<2048, 256, 0, stream>>>(x,  X16,  ((long)NN * EE) / 4);
    cvt_f32_f16<<<1024, 256, 0, stream>>>(qw, W16q, ((long)EE * EE) / 4);
    cvt_f32_f16<<<1024, 256, 0, stream>>>(kw, W16k, ((long)EE * EE) / 4);
    cvt_f32_f16<<<1024, 256, 0, stream>>>(vw, W16v, ((long)EE * EE) / 4);
    cvt_f32_f16<<<1024, 256, 0, stream>>>(ow, W16o, ((long)EE * EE) / 4);

    // QKV projections (WMMA GEMMs), 16x128 per wave
    dim3 gblock(32, 4);
    dim3 ggrid((MT + 3) / 4, EE / 128);
    gemm_proj<<<ggrid, gblock, 0, stream>>>(X16, W16q, qb, (void*)Q16,  0);
    gemm_proj<<<ggrid, gblock, 0, stream>>>(X16, W16k, kb, (void*)K16,  1);
    gemm_proj<<<ggrid, gblock, 0, stream>>>(X16, W16v, vb, (void*)VT16, 2);

    // attention: one wave per (head, 16-query tile)
    dim3 agrid(SPQ / 16, BB * HH);
    attn_kernel<<<agrid, dim3(32), 0, stream>>>(Q16, K16, VT16, CTX);

    // output projection -> f32 d_out [B,S,E]
    gemm_proj<<<ggrid, gblock, 0, stream>>>(CTX, W16o, ob, d_out, 3);
}
